// MultilayerGRU_56659208568876
// MI455X (gfx1250) — compile-verified, matrix operand order checked by reading, not run on hardware
//
#include <hip/hip_runtime.h>

// MI455X / gfx1250, wave32, WMMA bf16 path + async global->LDS pipeline.
typedef __bf16 bf;
typedef __attribute__((ext_vector_type(16))) __bf16 v16bf;
typedef __attribute__((ext_vector_type(8)))  float  v8f;
typedef int v4i __attribute__((vector_size(16)));   // matches builtin proto

#define GRU_B 128
#define GRU_S 512
#define GRU_H 1024
#define GRU_L 2
#define GRU_O 1024

#define AS1 __attribute__((address_space(1)))
#define AS3 __attribute__((address_space(3)))

#if defined(__HIP_DEVICE_COMPILE__) && \
    __has_builtin(__builtin_amdgcn_global_load_async_to_lds_b128)
#define HAVE_ASYNC_LDS 1
#else
#define HAVE_ASYNC_LDS 0
#endif

#if defined(__HIP_DEVICE_COMPILE__) && \
    __has_builtin(__builtin_amdgcn_s_wait_asynccnt)
#define WAIT_ASYNC(n) __builtin_amdgcn_s_wait_asynccnt(n)
#else
#define WAIT_ASYNC(n) asm volatile("s_wait_asynccnt %0" ::"i"(n) : "memory")
#endif

__device__ __forceinline__ bf f2bf(float f) {
  unsigned u = __builtin_bit_cast(unsigned, f);
  unsigned r = (u + 0x7FFFu + ((u >> 16) & 1u)) >> 16;  // round-to-nearest-even
  unsigned short s = (unsigned short)r;
  return __builtin_bit_cast(bf, s);
}

union ABFrag { v16bf v; uint4 u[2]; };

#if HAVE_ASYNC_LDS
__device__ __forceinline__ AS3 v4i* lds_v4(void* p) {
  return (AS3 v4i*)(uintptr_t)p;   // generic->LDS: low 32 bits are LDS addr
}
__device__ __forceinline__ AS1 v4i* glb_v4(const void* p) {
  return (AS1 v4i*)(uintptr_t)p;   // builtin takes non-const AS1 v4i*
}
// 64B contiguous per thread, both sides; IOFFSET applies to global AND LDS.
__device__ __forceinline__ void async_copy64(const bf* g, void* l) {
  __builtin_amdgcn_global_load_async_to_lds_b128(glb_v4(g), lds_v4(l), 0, 0);
  __builtin_amdgcn_global_load_async_to_lds_b128(glb_v4(g), lds_v4(l), 16, 0);
  __builtin_amdgcn_global_load_async_to_lds_b128(glb_v4(g), lds_v4(l), 32, 0);
  __builtin_amdgcn_global_load_async_to_lds_b128(glb_v4(g), lds_v4(l), 48, 0);
}
#endif

// ---------------------------------------------------------------------------
// GEMM: C[128, N] = A[128, K] * B[N, K]^T  (A, B bf16 row-major, acc f32)
// grid = (N/64, 128/64), block = 128 threads (4 waves); wave tile 32x32.
// Double-buffered LDS; async global->LDS copies overlap WMMA compute.
// MODE 0: z/r gates (sigmoid; z -> zbuf, r*h -> bf16 into rhdst second half)
// MODE 1: g gate + GRU update (tanh; h_new -> hout f32 and bf16 to d1/d2/d3)
// MODE 2: output projection (bias add, f32 store to yout)
// ---------------------------------------------------------------------------
template <int MODE>
__global__ __launch_bounds__(128) void gru_gemm(
    const bf* __restrict__ A, const bf* __restrict__ Bm, int K,
    const float* __restrict__ bias,
    float* __restrict__ zbuf, const float* __restrict__ hp,
    float* __restrict__ hout, bf* __restrict__ rhdst,
    bf* __restrict__ d1, int s1, bf* __restrict__ d2, int s2,
    bf* __restrict__ d3, int s3,
    float* __restrict__ yout, long ldy)
{
  __shared__ __align__(16) bf As[2][64][72];
  __shared__ __align__(16) bf Bs[2][64][72];

  const int tid   = threadIdx.x;
  const int lane  = tid & 31;
  const int wave  = tid >> 5;
  const int wm    = wave >> 1;        // 0..1 : M sub-tile of WG tile
  const int wn    = wave & 1;         // 0..1 : N sub-tile of WG tile
  const int M0    = blockIdx.y * 64;
  const int N0    = blockIdx.x * 64;
  const int lrow  = tid >> 1;         // 0..63, row staged by this thread
  const int lseg  = (tid & 1) * 32;   // K half (32 bf16 = 64B)
  const int khalf = lane >> 4;        // fragment: lanes 16-31 hold K+8 chunk
  const int ml    = lane & 15;        // fragment row (M for A, N for B)

  v8f acc[2][2] = {};

  const bf* aptr = A  + (long)(M0 + lrow) * K + lseg;
  const bf* bptr = Bm + (long)(N0 + lrow) * K + lseg;

  // 8 WMMAs on LDS stage st (ISA 7.12.2 16-bit A/B fragment layout: two
  // contiguous 8-elem K-chunks per lane at kk+khalf*8 and kk+16+khalf*8).
  auto compute_stage = [&](int st) {
#pragma unroll
    for (int kk = 0; kk < 64; kk += 32) {
      ABFrag fa[2], fb[2];
#pragma unroll
      for (int mi = 0; mi < 2; ++mi) {
        int m = wm * 32 + mi * 16 + ml;
        fa[mi].u[0] = *(const uint4*)&As[st][m][kk + khalf * 8];
        fa[mi].u[1] = *(const uint4*)&As[st][m][kk + 16 + khalf * 8];
      }
#pragma unroll
      for (int ni = 0; ni < 2; ++ni) {
        int n = wn * 32 + ni * 16 + ml;
        fb[ni].u[0] = *(const uint4*)&Bs[st][n][kk + khalf * 8];
        fb[ni].u[1] = *(const uint4*)&Bs[st][n][kk + 16 + khalf * 8];
      }
#pragma unroll
      for (int mi = 0; mi < 2; ++mi)
#pragma unroll
        for (int ni = 0; ni < 2; ++ni)
          acc[mi][ni] = __builtin_amdgcn_wmma_f32_16x16x32_bf16(
              false, fa[mi].v, false, fb[ni].v, (short)0, acc[mi][ni],
              false, false);
    }
  };

#if HAVE_ASYNC_LDS
  // -------- async double-buffered pipeline (ASYNCcnt) --------
  async_copy64(aptr, &As[0][lrow][lseg]);            // prologue: stage 0
  async_copy64(bptr, &Bs[0][lrow][lseg]);
  for (int k0 = 0; k0 < K; k0 += 64) {
    const int st = (k0 >> 6) & 1;
    __syncthreads();                 // all waves done computing stage st^1
    if (k0 + 64 < K) {
      async_copy64(aptr + 64, &As[st ^ 1][lrow][lseg]);
      async_copy64(bptr + 64, &Bs[st ^ 1][lrow][lseg]);
      WAIT_ASYNC(8);                 // in-order: batch for stage st landed
    } else {
      WAIT_ASYNC(0);
    }
    aptr += 64; bptr += 64;
    __syncthreads();                 // stage st visible to all waves
    compute_stage(st);
  }
#else
  // -------- fallback: synchronous staging through VGPRs --------
  for (int k0 = 0; k0 < K; k0 += 64) {
    uint4 a0 = *(const uint4*)(aptr +  0);
    uint4 a1 = *(const uint4*)(aptr +  8);
    uint4 a2 = *(const uint4*)(aptr + 16);
    uint4 a3 = *(const uint4*)(aptr + 24);
    uint4 b0 = *(const uint4*)(bptr +  0);
    uint4 b1 = *(const uint4*)(bptr +  8);
    uint4 b2 = *(const uint4*)(bptr + 16);
    uint4 b3 = *(const uint4*)(bptr + 24);
    aptr += 64; bptr += 64;
    if (k0 + 64 < K) {
      __builtin_prefetch(aptr, 0, 1);
      __builtin_prefetch(bptr, 0, 1);
    }
    __syncthreads();
    *(uint4*)&As[0][lrow][lseg +  0] = a0;
    *(uint4*)&As[0][lrow][lseg +  8] = a1;
    *(uint4*)&As[0][lrow][lseg + 16] = a2;
    *(uint4*)&As[0][lrow][lseg + 24] = a3;
    *(uint4*)&Bs[0][lrow][lseg +  0] = b0;
    *(uint4*)&Bs[0][lrow][lseg +  8] = b1;
    *(uint4*)&Bs[0][lrow][lseg + 16] = b2;
    *(uint4*)&Bs[0][lrow][lseg + 24] = b3;
    __syncthreads();
    compute_stage(0);
  }
#endif

  // Epilogue. C/D layout: VGPR i -> (M = i + khalf*8, N = ml).
#pragma unroll
  for (int mi = 0; mi < 2; ++mi) {
#pragma unroll
    for (int ni = 0; ni < 2; ++ni) {
#pragma unroll
      for (int i = 0; i < 8; ++i) {
        int row = M0 + wm * 32 + mi * 16 + khalf * 8 + i;
        int col = N0 + wn * 32 + ni * 16 + ml;
        float c = acc[mi][ni][i];
        if (MODE == 0) {
          float sg = 1.0f / (1.0f + __expf(-(c + bias[col])));
          if (col < GRU_H) {
            zbuf[row * GRU_H + col] = sg;                       // z gate
          } else {
            int h = col - GRU_H;                                // r gate
            float rh = sg * hp[row * GRU_H + h];
            rhdst[row * (2 * GRU_H) + GRU_H + h] = f2bf(rh);    // A of g-GEMM
          }
        } else if (MODE == 1) {
          float g = tanhf(c + bias[col]);
          float z = zbuf[row * GRU_H + col];
          float hprev = hp[row * GRU_H + col];
          float hn = z * hprev + (1.0f - z) * g;
          hout[row * GRU_H + col] = hn;                         // f32 master
          bf hb = f2bf(hn);
          d1[row * s1 + col] = hb;
          d2[row * s2 + col] = hb;
          if (d3) d3[row * s3 + col] = hb;
        } else {
          yout[(long)row * ldy + col] = c + bias[col];          // y = hWhy^T+b
        }
      }
    }
  }
}

// --------------------------- helper kernels --------------------------------

// a_zr[:,0:H] = a_g[:,0:H] = bf16(x[:,t,:])     (layer 0 input per step)
__global__ void pack_x_kernel(const float* __restrict__ x,
                              bf* __restrict__ a_zr, bf* __restrict__ a_g,
                              int t) {
  int i = blockIdx.x * 256 + threadIdx.x;   // over B*H
  int b = i >> 10, h = i & 1023;
  bf v = f2bf(x[((long)b * GRU_S + t) * GRU_H + h]);
  a_zr[b * 2048 + h] = v;
  a_g [b * 2048 + h] = v;
}

// Wzr[l][n][k] bf16, n in [0,2048): gate = n>>10 (z,r), k<H from Wx, else Wh.
__global__ void prep_wzr_kernel(const float* __restrict__ Wx,
                                const float* __restrict__ Wh,
                                bf* __restrict__ Wzr) {
  int i = blockIdx.x * 256 + threadIdx.x;   // over L*2048*2048
  int k = i & 2047;
  int n = (i >> 11) & 2047;
  int l = i >> 22;
  int gate = n >> 10, nn = n & 1023;
  float v = (k < 1024)
      ? Wx[(((long)l * 3 + gate) * 1024 + nn) * 1024 + k]
      : Wh[(((long)l * 3 + gate) * 1024 + nn) * 1024 + (k - 1024)];
  Wzr[i] = f2bf(v);
}

// Wg[l][n][k] bf16, gate g: k<H from Wx[l,2], else Wh[l,2].
__global__ void prep_wg_kernel(const float* __restrict__ Wx,
                               const float* __restrict__ Wh,
                               bf* __restrict__ Wg) {
  int i = blockIdx.x * 256 + threadIdx.x;   // over L*1024*2048
  int k = i & 2047;
  int n = (i >> 11) & 1023;
  int l = i >> 21;
  float v = (k < 1024)
      ? Wx[(((long)l * 3 + 2) * 1024 + n) * 1024 + k]
      : Wh[(((long)l * 3 + 2) * 1024 + n) * 1024 + (k - 1024)];
  Wg[i] = f2bf(v);
}

__global__ void prep_why_kernel(const float* __restrict__ Why,
                                bf* __restrict__ Whyb) {
  int i = blockIdx.x * 256 + threadIdx.x;   // over O*H
  Whyb[i] = f2bf(Why[i]);
}

// h := 0 (both layers) and zero the h-halves of the zr A-operands.
__global__ void init_kernel(float* __restrict__ hf,
                            bf* __restrict__ a0_zr, bf* __restrict__ a1_zr) {
  int i = blockIdx.x * 256 + threadIdx.x;   // over B*H
  int b = i >> 10, h = i & 1023;
  hf[i] = 0.0f;
  hf[GRU_B * GRU_H + i] = 0.0f;
  bf z = f2bf(0.0f);
  a0_zr[b * 2048 + 1024 + h] = z;
  a1_zr[b * 2048 + 1024 + h] = z;
}

// hidden_state output: out2[b][l][h] = h_f32[l][b][h]
__global__ void final_hidden_kernel(const float* __restrict__ hf,
                                    float* __restrict__ out) {
  int i = blockIdx.x * 256 + threadIdx.x;   // over L*B*H
  int l = i >> 17;
  int r = i & ((1 << 17) - 1);
  int b = r >> 10, h = r & 1023;
  out[(long)GRU_B * GRU_S * GRU_O + ((long)b * GRU_L + l) * GRU_H + h] = hf[i];
}

// ---------------------------------------------------------------------------

extern "C" void kernel_launch(void* const* d_in, const int* in_sizes, int n_in,
                              void* d_out, int out_size, void* d_ws,
                              size_t ws_size, hipStream_t stream) {
  (void)in_sizes; (void)n_in; (void)out_size; (void)ws_size;
  const float* x   = (const float*)d_in[0];
  const float* Wx  = (const float*)d_in[1];
  const float* Wh  = (const float*)d_in[2];
  const float* bh  = (const float*)d_in[3];
  const float* Why = (const float*)d_in[4];
  const float* by  = (const float*)d_in[5];
  float* out = (float*)d_out;

  char* p = (char*)d_ws;
  auto alloc = [&](size_t bytes) -> char* {
    char* r = p; p += (bytes + 255) & ~(size_t)255; return r;
  };

  bf* Wzr    = (bf*)alloc((size_t)GRU_L * 2048 * 2048 * sizeof(bf)); // 16 MB
  bf* Wg     = (bf*)alloc((size_t)GRU_L * 1024 * 2048 * sizeof(bf)); //  8 MB
  bf* Whyb   = (bf*)alloc((size_t)1024 * 1024 * sizeof(bf));         //  2 MB
  bf* a0_zr  = (bf*)alloc((size_t)128 * 2048 * sizeof(bf));
  bf* a0_g   = (bf*)alloc((size_t)128 * 2048 * sizeof(bf));
  bf* a1_zr  = (bf*)alloc((size_t)128 * 2048 * sizeof(bf));
  bf* a1_g   = (bf*)alloc((size_t)128 * 2048 * sizeof(bf));
  bf* a_y    = (bf*)alloc((size_t)128 * 1024 * sizeof(bf));
  float* hf  = (float*)alloc((size_t)GRU_L * 128 * 1024 * sizeof(float));
  float* zb0 = (float*)alloc((size_t)128 * 1024 * sizeof(float));
  float* zb1 = (float*)alloc((size_t)128 * 1024 * sizeof(float));
  float* hf1 = hf + 128 * 1024;

  // one-time (per call) weight conversion + state init
  prep_wzr_kernel<<<(GRU_L * 2048 * 2048) / 256, 256, 0, stream>>>(Wx, Wh, Wzr);
  prep_wg_kernel <<<(GRU_L * 1024 * 2048) / 256, 256, 0, stream>>>(Wx, Wh, Wg);
  prep_why_kernel<<<(1024 * 1024) / 256, 256, 0, stream>>>(Why, Whyb);
  init_kernel    <<<(128 * 1024) / 256, 256, 0, stream>>>(hf, a0_zr, a1_zr);

  const dim3 blk(128);
  const float* bh0 = bh;                    // layer 0: z,r rows then g
  const float* bh1 = bh + 3 * GRU_H;        // layer 1
  bf* Wzr1 = Wzr + (size_t)2048 * 2048;
  bf* Wg1  = Wg  + (size_t)1024 * 2048;

  for (int t = 0; t < GRU_S; ++t) {
    pack_x_kernel<<<(128 * 1024) / 256, 256, 0, stream>>>(x, a0_zr, a0_g, t);

    // ---- layer 0 ----
    gru_gemm<0><<<dim3(2048 / 64, 2), blk, 0, stream>>>(
        a0_zr, Wzr, 2048, bh0, zb0, hf, nullptr, a0_g,
        nullptr, 0, nullptr, 0, nullptr, 0, nullptr, 0);
    gru_gemm<1><<<dim3(1024 / 64, 2), blk, 0, stream>>>(
        a0_g, Wg, 2048, bh0 + 2 * GRU_H, zb0, hf, hf, nullptr,
        a1_zr, 2048,            // layer-1 zr input (first half)
        a1_g, 2048,             // layer-1 g input (first half)
        a0_zr + 1024, 2048,     // own h half for next step
        nullptr, 0);

    // ---- layer 1 ----
    gru_gemm<0><<<dim3(2048 / 64, 2), blk, 0, stream>>>(
        a1_zr, Wzr1, 2048, bh1, zb1, hf1, nullptr, a1_g,
        nullptr, 0, nullptr, 0, nullptr, 0, nullptr, 0);
    gru_gemm<1><<<dim3(1024 / 64, 2), blk, 0, stream>>>(
        a1_g, Wg1, 2048, bh1 + 2 * GRU_H, zb1, hf1, hf1, nullptr,
        a_y, 1024,              // projection input
        a1_zr + 1024, 2048,     // own h half for next step
        nullptr, 0,
        nullptr, 0);

    // ---- output projection: y[:,t,:] = h1 @ Why^T + by ----
    gru_gemm<2><<<dim3(1024 / 64, 2), blk, 0, stream>>>(
        a_y, Whyb, 1024, by, nullptr, nullptr, nullptr, nullptr,
        nullptr, 0, nullptr, 0, nullptr, 0,
        out + (long)t * GRU_O, (long)GRU_S * GRU_O);
  }

  final_hidden_kernel<<<(GRU_L * 128 * 1024) / 256, 256, 0, stream>>>(hf, out);
}
